// Estor_attention_45595372814582
// MI455X (gfx1250) — matrix operand order repaired and checked by reading, not compiled, not run on hardware
//
#include <hip/hip_runtime.h>

// ---------------------------------------------------------------------------
// Types for CDNA5 WMMA (wave32, v_wmma_f32_16x16x32_bf16)
// ---------------------------------------------------------------------------
typedef __bf16 bf16_t;
typedef bf16_t v16bf __attribute__((ext_vector_type(16)));
typedef float  v8f   __attribute__((ext_vector_type(8)));

union FragU { uint4 u[2]; v16bf v; };

// Problem constants
#define BB   8
#define SS   2048
#define HH   1024
#define NHH  16
#define DHH  64
#define FF   4096
#define NSPAN 512
#define LMAX  64
#define TT    65          // 1 tag token + LMAX span tokens
#define MROWS (NSPAN*TT)  // 33280
#define NTOK  (BB*SS)     // 16384

// ---------------------------------------------------------------------------
// CDNA5 async global->LDS DMA helpers (ASYNCcnt-tracked, CDNA5 ISA)
// Moves 32 bytes per lane: INST_OFFSET applies to BOTH LDS and global address.
// ---------------------------------------------------------------------------
__device__ __forceinline__ unsigned lds_addr32(const void* p) {
    // Generic (flat) pointers into the LDS aperture carry the LDS byte offset
    // in their low 32 bits (ISA 10.2).
    return (unsigned)(size_t)p;
}
__device__ __forceinline__ void async_b128x2(unsigned lds, const void* g) {
    asm volatile("global_load_async_to_lds_b128 %0, %1, off\n\t"
                 "global_load_async_to_lds_b128 %0, %1, off offset:16"
                 :: "v"(lds), "v"(g) : "memory");
}
__device__ __forceinline__ void wait_async0() {
    asm volatile("s_wait_asynccnt 0" ::: "memory");
}

// ---------------------------------------------------------------------------
// f32 -> bf16 conversion (weights)
// ---------------------------------------------------------------------------
__global__ __launch_bounds__(256) void cvt_bf16_kernel(const float* __restrict__ src,
                                                       bf16_t* __restrict__ dst, int n) {
    int i = blockIdx.x * 256 + threadIdx.x;
    if (i < n) dst[i] = (bf16_t)src[i];
}

// ---------------------------------------------------------------------------
// Fused RoPE + span gather + tag row -> Q bf16 (MROWS x H)
// ---------------------------------------------------------------------------
__global__ __launch_bounds__(256) void build_q_kernel(
    const float* __restrict__ word, const int* __restrict__ span_batch,
    const int* __restrict__ span_tag, const int* __restrict__ span_start,
    const float* __restrict__ tag_emb, bf16_t* __restrict__ Q)
{
    const int s = blockIdx.x;           // span
    const int t = blockIdx.y;           // 0..64
    const size_t orow = ((size_t)s * TT + t) * HH;
    if (t == 0) {
        const float* src = tag_emb + (size_t)span_tag[s] * HH;
        for (int i = threadIdx.x; i < HH; i += 256)
            Q[orow + i] = (bf16_t)src[i];
    } else {
        const int b = span_batch[s];
        const int p = span_start[s] + (t - 1);       // absolute sequence position
        const float* x = word + ((size_t)b * SS + p) * HH;
        const float L2_10000 = 13.287712379549449f;  // log2(10000)
        for (int i = threadIdx.x; i < HH / 2; i += 256) {
            float inv = exp2f(-L2_10000 * (float)(2 * i) / (float)HH);
            float ang = (float)p * inv;
            float sn, cs;
            __sincosf(ang, &sn, &cs);
            float x0 = x[2 * i], x1 = x[2 * i + 1];
            Q[orow + 2 * i]     = (bf16_t)(x0 * cs - x1 * sn);
            Q[orow + 2 * i + 1] = (bf16_t)(x1 * cs + x0 * sn);
        }
    }
}

// ---------------------------------------------------------------------------
// Tiled bf16 WMMA GEMM:  C(MxN) = A(MxK) @ W(NxK)^T + bias
// Block tile 128x128, 8 waves (wave32), each wave 32x64 = 2x4 WMMA tiles.
// Double-buffered LDS, tiles staged by GLOBAL_LOAD_ASYNC_TO_LDS_B128 so the
// next tile's DMA overlaps the current tile's WMMAs; one barrier per K-step.
// M,N multiples of 128; K multiple of 32.
// ---------------------------------------------------------------------------
template <bool RELU, bool OUT_BF16>
__global__ __launch_bounds__(256) void gemm_bf16_kernel(
    const bf16_t* __restrict__ A, const bf16_t* __restrict__ W,
    const float* __restrict__ bias, float* __restrict__ Cf,
    bf16_t* __restrict__ Cb, int M, int N, int K)
{
    // Row stride 40 bf16 = 80 bytes (multiple of 16B for b128 LDS access)
    __shared__ bf16_t As[2][128][40];
    __shared__ bf16_t Ws[2][128][40];

    const int tid  = threadIdx.x;
    const int wave = tid >> 5;
    const int lane = tid & 31;
    const int wm   = wave >> 1;   // 0..3  (rows wm*32)
    const int wn   = wave & 1;    // 0..1  (cols wn*64)
    const int half = lane >> 4;   // lane half
    const int lm   = lane & 15;

    const int m0 = blockIdx.y * 128;
    const int n0 = blockIdx.x * 128;

    const int lrow = tid >> 1;          // 0..127
    const int lseg = (tid & 1) * 16;    // 0 or 16 (bf16 elements)

    // Per-thread staging addresses: 32B global segment -> 32B LDS segment
    const bf16_t* gA = A + (size_t)(m0 + lrow) * K + lseg;
    const bf16_t* gW = W + (size_t)(n0 + lrow) * K + lseg;
    const unsigned lA[2] = { lds_addr32(&As[0][lrow][lseg]), lds_addr32(&As[1][lrow][lseg]) };
    const unsigned lW[2] = { lds_addr32(&Ws[0][lrow][lseg]), lds_addr32(&Ws[1][lrow][lseg]) };

    v8f acc[2][4] = {};

    const int nIter = K >> 5;

    // Prologue: DMA tile 0 into buffer 0
    async_b128x2(lA[0], gA);
    async_b128x2(lW[0], gW);

    for (int it = 0; it < nIter; ++it) {
        const int cur = it & 1;

        wait_async0();       // my async writes into buf[cur] have landed
        __syncthreads();     // all waves landed theirs; all reads of buf[cur^1] done
                             // (barrier lowering includes s_wait_dscnt 0)

        if (it + 1 < nIter) {  // overlap next tile's DMA with this tile's math
            const bf16_t* gA2 = gA + (size_t)(it + 1) * 32;
            const bf16_t* gW2 = gW + (size_t)(it + 1) * 32;
            async_b128x2(lA[cur ^ 1], gA2);
            async_b128x2(lW[cur ^ 1], gW2);
        }

        // A fragment (16x32): lane half 0 -> K {0..7,16..23}, half 1 -> K {8..15,24..31}
        FragU fa[2], fb[4];
#pragma unroll
        for (int mt = 0; mt < 2; ++mt) {
            const uint4* ap = reinterpret_cast<const uint4*>(
                &As[cur][wm * 32 + mt * 16 + lm][half * 8]);
            fa[mt].u[0] = ap[0];
            fa[mt].u[1] = ap[2];   // +16 elements
        }
        // B fragment (32x16): lane half 0 -> K {0..7,16..23}, half 1 -> K {8..15,24..31}
#pragma unroll
        for (int nt = 0; nt < 4; ++nt) {
            const uint4* bp = reinterpret_cast<const uint4*>(
                &Ws[cur][wn * 64 + nt * 16 + lm][half * 8]);
            fb[nt].u[0] = bp[0];
            fb[nt].u[1] = bp[2];
        }
#pragma unroll
        for (int mt = 0; mt < 2; ++mt)
#pragma unroll
            for (int nt = 0; nt < 4; ++nt)
                acc[mt][nt] = __builtin_amdgcn_wmma_f32_16x16x32_bf16(
                    false, fa[mt].v, false, fb[nt].v,
                    (short)0, acc[mt][nt], false, false);
    }

    // Epilogue: C/D layout: lane half h, VGPR j -> row = 8h + j, col = lm
#pragma unroll
    for (int mt = 0; mt < 2; ++mt) {
#pragma unroll
        for (int nt = 0; nt < 4; ++nt) {
            const int col = n0 + wn * 64 + nt * 16 + lm;
            const float bv = bias[col];
#pragma unroll
            for (int j = 0; j < 8; ++j) {
                const int row = m0 + wm * 32 + mt * 16 + half * 8 + j;
                float val = acc[mt][nt][j] + bv;
                if (RELU) val = fmaxf(val, 0.f);
                if (OUT_BF16) Cb[(size_t)row * N + col] = (bf16_t)val;
                else          Cf[(size_t)row * N + col] = val;
            }
        }
    }
}

// ---------------------------------------------------------------------------
// Per-(span, head) attention: T=65, DH=64, f32 math from bf16 QKV
// ---------------------------------------------------------------------------
__global__ __launch_bounds__(256) void attn_kernel(
    const bf16_t* __restrict__ qkv, const int* __restrict__ span_len,
    bf16_t* __restrict__ O)
{
    const int s = blockIdx.x;   // span
    const int h = blockIdx.y;   // head

    __shared__ float Qs[TT][DHH];
    __shared__ float Ks[TT][DHH + 1];  // pad: lane-distinct rows -> distinct banks
    __shared__ float Vs[TT][DHH];

    const int len = span_len[s];
    for (int i = threadIdx.x; i < TT * DHH; i += 256) {
        int t = i >> 6, d = i & 63;
        size_t base = ((size_t)(s * TT + t)) * (3 * HH) + h * DHH + d;
        Qs[t][d] = (float)qkv[base];
        Ks[t][d] = (float)qkv[base + HH];
        Vs[t][d] = (float)qkv[base + 2 * HH];
    }
    __syncthreads();

    const int wave = threadIdx.x >> 5;
    const int lane = threadIdx.x & 31;

    for (int t = wave; t < TT; t += 8) {
        // scores for keys owned by this lane: j = lane, lane+32, lane+64
        float sc[3];
#pragma unroll
        for (int jj = 0; jj < 3; ++jj) {
            int j = jj * 32 + lane;
            float v = -1e30f;
            if (j < TT) {
                bool valid = (j == 0) || ((j - 1) < len);
                if (valid) {
                    float dot = 0.f;
#pragma unroll
                    for (int d = 0; d < DHH; ++d) dot += Qs[t][d] * Ks[j][d];
                    v = dot * 0.125f;   // 1/sqrt(64)
                }
            }
            sc[jj] = v;
        }
        float m = fmaxf(sc[0], fmaxf(sc[1], sc[2]));
        for (int o = 16; o; o >>= 1) m = fmaxf(m, __shfl_xor(m, o, 32));
        float pj[3];
        float sum = 0.f;
#pragma unroll
        for (int jj = 0; jj < 3; ++jj) {
            pj[jj] = (sc[jj] > -1e29f) ? __expf(sc[jj] - m) : 0.f;
            sum += pj[jj];
        }
        for (int o = 16; o; o >>= 1) sum += __shfl_xor(sum, o, 32);
        float inv = 1.f / sum;
#pragma unroll
        for (int jj = 0; jj < 3; ++jj) pj[jj] *= inv;

        // o_d = sum_j p_j * V[j][d]; broadcast p_j via shuffles
        float o0 = 0.f, o1 = 0.f;
#pragma unroll
        for (int j = 0; j < TT; ++j) {
            float p = __shfl(pj[j >> 5], j & 31, 32);
            o0 += p * Vs[j][lane];
            o1 += p * Vs[j][lane + 32];
        }
        size_t orow = ((size_t)(s * TT + t)) * HH + h * DHH;
        O[orow + lane]      = (bf16_t)o0;
        O[orow + lane + 32] = (bf16_t)o1;
    }
}

// ---------------------------------------------------------------------------
// acc = word_embedding (residual base)
// ---------------------------------------------------------------------------
__global__ __launch_bounds__(256) void copy_f32_kernel(const float* __restrict__ src,
                                                       float* __restrict__ dst, size_t n) {
    size_t i = (size_t)blockIdx.x * 256 + threadIdx.x;
    if (i < n) dst[i] = src[i];
}

// ---------------------------------------------------------------------------
// Scatter-add span outputs: acc[b, pos, :] += 0.5 * att[row, :]
// ---------------------------------------------------------------------------
__global__ __launch_bounds__(256) void scatter_kernel(
    const float* __restrict__ att, const int* __restrict__ span_batch,
    const int* __restrict__ span_start, const int* __restrict__ span_len,
    float* __restrict__ acc)
{
    const int s = blockIdx.x;
    const int t = blockIdx.y;            // token within span, 0..63
    if (t >= span_len[s]) return;
    const int b   = span_batch[s];
    const int pos = span_start[s] + t;
    const float* src = att + ((size_t)(s * TT) + 1 + t) * HH;
    float* dst = acc + ((size_t)b * SS + pos) * HH;
    for (int c = threadIdx.x; c < HH; c += 256)
        atomicAdd(&dst[c], 0.5f * src[c]);   // TAGGING_RATE = 0.5
}

// ---------------------------------------------------------------------------
// LayerNorm over H=1024 per token -> f32 + bf16 outputs
// ---------------------------------------------------------------------------
__global__ __launch_bounds__(256) void ln_kernel(
    const float* __restrict__ x, const float* __restrict__ g,
    const float* __restrict__ b, float* __restrict__ yf, bf16_t* __restrict__ yb)
{
    const int tok = blockIdx.x;
    const float* xr = x + (size_t)tok * HH;
    float v[4];
    float s = 0.f, s2 = 0.f;
#pragma unroll
    for (int i = 0; i < 4; ++i) {
        int idx = threadIdx.x + i * 256;
        v[i] = xr[idx];
        s += v[i]; s2 += v[i] * v[i];
    }
    for (int o = 16; o; o >>= 1) { s += __shfl_xor(s, o, 32); s2 += __shfl_xor(s2, o, 32); }
    __shared__ float rs[8], rs2[8];
    const int wave = threadIdx.x >> 5, lane = threadIdx.x & 31;
    if (lane == 0) { rs[wave] = s; rs2[wave] = s2; }
    __syncthreads();
    float S = 0.f, S2 = 0.f;
#pragma unroll
    for (int w = 0; w < 8; ++w) { S += rs[w]; S2 += rs2[w]; }
    float mean = S * (1.f / HH);
    float var  = S2 * (1.f / HH) - mean * mean;
    float inv  = rsqrtf(var + 1e-12f);
#pragma unroll
    for (int i = 0; i < 4; ++i) {
        int idx = threadIdx.x + i * 256;
        float n = (v[i] - mean) * inv * g[idx] + b[idx];
        yf[(size_t)tok * HH + idx] = n;
        yb[(size_t)tok * HH + idx] = (bf16_t)n;
    }
}

// ---------------------------------------------------------------------------
// Fused: y = ff2 + ln1 ; LN2(y) ; out = LN2 @ out_w^T + out_b (9 labels)
// ---------------------------------------------------------------------------
__global__ __launch_bounds__(256) void final_kernel(
    const float* __restrict__ ff2, const float* __restrict__ res,
    const float* __restrict__ g, const float* __restrict__ b,
    const float* __restrict__ ow, const float* __restrict__ ob,
    float* __restrict__ out)
{
    const int tok = blockIdx.x;
    __shared__ float norm[HH];
    __shared__ float rs[8], rs2[8], red[9];

    float v[4];
    float s = 0.f, s2 = 0.f;
#pragma unroll
    for (int i = 0; i < 4; ++i) {
        int idx = threadIdx.x + i * 256;
        float val = ff2[(size_t)tok * HH + idx] + res[(size_t)tok * HH + idx];
        v[i] = val; s += val; s2 += val * val;
    }
    for (int o = 16; o; o >>= 1) { s += __shfl_xor(s, o, 32); s2 += __shfl_xor(s2, o, 32); }
    const int wave = threadIdx.x >> 5, lane = threadIdx.x & 31;
    if (lane == 0) { rs[wave] = s; rs2[wave] = s2; }
    __syncthreads();
    float S = 0.f, S2 = 0.f;
#pragma unroll
    for (int w = 0; w < 8; ++w) { S += rs[w]; S2 += rs2[w]; }
    float mean = S * (1.f / HH);
    float var  = S2 * (1.f / HH) - mean * mean;
    float inv  = rsqrtf(var + 1e-12f);
#pragma unroll
    for (int i = 0; i < 4; ++i) {
        int idx = threadIdx.x + i * 256;
        norm[idx] = (v[i] - mean) * inv * g[idx] + b[idx];
    }
    if (threadIdx.x < 9) red[threadIdx.x] = 0.f;
    __syncthreads();

    float part[9] = {0.f, 0.f, 0.f, 0.f, 0.f, 0.f, 0.f, 0.f, 0.f};
    for (int hh = threadIdx.x; hh < HH; hh += 256) {
        float n = norm[hh];
#pragma unroll
        for (int l = 0; l < 9; ++l) part[l] += n * ow[l * HH + hh];
    }
#pragma unroll
    for (int l = 0; l < 9; ++l) {
        float p = part[l];
        for (int o = 16; o; o >>= 1) p += __shfl_xor(p, o, 32);
        if (lane == 0) atomicAdd(&red[l], p);
    }
    __syncthreads();
    if (threadIdx.x < 9)
        out[(size_t)tok * 9 + threadIdx.x] = red[threadIdx.x] + ob[threadIdx.x];
}

// ---------------------------------------------------------------------------
// Host-side orchestration
// ---------------------------------------------------------------------------
extern "C" void kernel_launch(void* const* d_in, const int* in_sizes, int n_in,
                              void* d_out, int out_size, void* d_ws, size_t ws_size,
                              hipStream_t stream) {
    const float* word       = (const float*)d_in[0];
    const int*   span_batch = (const int*)d_in[1];
    const int*   span_tag   = (const int*)d_in[2];
    const int*   span_start = (const int*)d_in[3];
    const int*   span_len   = (const int*)d_in[4];
    const float* tag_emb    = (const float*)d_in[5];
    const float* in_w  = (const float*)d_in[6];
    const float* in_b  = (const float*)d_in[7];
    const float* opw   = (const float*)d_in[8];
    const float* opb   = (const float*)d_in[9];
    const float* g1    = (const float*)d_in[10];
    const float* bn1   = (const float*)d_in[11];
    const float* f1w   = (const float*)d_in[12];
    const float* f1b   = (const float*)d_in[13];
    const float* f2w   = (const float*)d_in[14];
    const float* f2b   = (const float*)d_in[15];
    const float* g2    = (const float*)d_in[16];
    const float* bn2   = (const float*)d_in[17];
    const float* ow    = (const float*)d_in[18];
    const float* ob    = (const float*)d_in[19];
    float* out = (float*)d_out;

    char* ws = (char*)d_ws;
    size_t off = 0;
    auto alloc = [&](size_t bytes) {
        size_t o = off;
        off = (off + bytes + 255) & ~(size_t)255;
        return o;
    };

    // Weight bf16 copies (persistent)
    size_t oWIn  = alloc((size_t)3 * HH * HH * 2);
    size_t oWOut = alloc((size_t)HH * HH * 2);
    size_t oWF1  = alloc((size_t)FF * HH * 2);
    size_t oWF2  = alloc((size_t)HH * FF * 2);
    // Region B: Q bf16, later reused for attention output O bf16 (same size)
    size_t oQ    = alloc((size_t)MROWS * HH * 2);
    // Region A: QKV bf16 (204 MB), later reused for FFN hidden H bf16 (134 MB)
    size_t oA    = alloc((size_t)MROWS * 3 * HH * 2);
    // Region C: ATT f32 (136 MB), later [LN1 f32 (64 MB) | LN1 bf16 (34 MB)]
    size_t oC    = alloc((size_t)MROWS * HH * 4);
    // Region D: ACC f32 (64 MB), later FF2 f32 (64 MB)
    size_t oD    = alloc((size_t)NTOK * HH * 4);
    (void)ws_size; (void)n_in; (void)in_sizes; (void)out_size;

    bf16_t* wInB  = (bf16_t*)(ws + oWIn);
    bf16_t* wOutB = (bf16_t*)(ws + oWOut);
    bf16_t* wF1B  = (bf16_t*)(ws + oWF1);
    bf16_t* wF2B  = (bf16_t*)(ws + oWF2);
    bf16_t* Qb    = (bf16_t*)(ws + oQ);
    bf16_t* Ob    = (bf16_t*)(ws + oQ);   // alias: Q dead after GEMM1
    bf16_t* QKV   = (bf16_t*)(ws + oA);
    bf16_t* Hb    = (bf16_t*)(ws + oA);   // alias: QKV dead after attention
    float*  ATT   = (float*)(ws + oC);
    float*  LN1F  = (float*)(ws + oC);    // alias: ATT dead after scatter
    bf16_t* LN1B  = (bf16_t*)(ws + oC + (size_t)NTOK * HH * 4);
    float*  ACC   = (float*)(ws + oD);
    float*  FF2   = (float*)(ws + oD);    // alias: ACC dead after LN1

    // 1) Convert weights to bf16
    cvt_bf16_kernel<<<(3 * HH * HH + 255) / 256, 256, 0, stream>>>(in_w, wInB, 3 * HH * HH);
    cvt_bf16_kernel<<<(HH * HH + 255) / 256, 256, 0, stream>>>(opw, wOutB, HH * HH);
    cvt_bf16_kernel<<<(FF * HH + 255) / 256, 256, 0, stream>>>(f1w, wF1B, FF * HH);
    cvt_bf16_kernel<<<(HH * FF + 255) / 256, 256, 0, stream>>>(f2w, wF2B, HH * FF);

    // 2) RoPE + gather span tokens + tag rows -> Q bf16
    build_q_kernel<<<dim3(NSPAN, TT), 256, 0, stream>>>(
        word, span_batch, span_tag, span_start, tag_emb, Qb);

    // 3) QKV projection: (33280 x 1024) @ (1024 x 3072) + b_in -> bf16
    gemm_bf16_kernel<false, true><<<dim3(3 * HH / 128, MROWS / 128), 256, 0, stream>>>(
        Qb, wInB, in_b, (float*)nullptr, QKV, MROWS, 3 * HH, HH);

    // 4) Per-(span, head) attention -> O bf16
    attn_kernel<<<dim3(NSPAN, NHH), 256, 0, stream>>>(QKV, span_len, Ob);

    // 5) Output projection: (33280 x 1024) @ (1024 x 1024) + b_out -> f32
    gemm_bf16_kernel<false, false><<<dim3(HH / 128, MROWS / 128), 256, 0, stream>>>(
        Ob, wOutB, opb, ATT, (bf16_t*)nullptr, MROWS, HH, HH);

    // 6) acc = word_embedding ; scatter-add 0.5 * span outputs
    copy_f32_kernel<<<(unsigned)(((size_t)NTOK * HH + 255) / 256), 256, 0, stream>>>(
        word, ACC, (size_t)NTOK * HH);
    scatter_kernel<<<dim3(NSPAN, LMAX), 256, 0, stream>>>(
        ATT, span_batch, span_start, span_len, ACC);

    // 7) LayerNorm 1 -> f32 + bf16
    ln_kernel<<<NTOK, 256, 0, stream>>>(ACC, g1, bn1, LN1F, LN1B);

    // 8) FFN GEMM1: (16384 x 1024) @ (1024 x 4096) + b1, ReLU -> bf16
    gemm_bf16_kernel<true, true><<<dim3(FF / 128, NTOK / 128), 256, 0, stream>>>(
        LN1B, wF1B, f1b, (float*)nullptr, Hb, NTOK, FF, HH);

    // 9) FFN GEMM2: (16384 x 4096) @ (4096 x 1024) + b2 -> f32
    gemm_bf16_kernel<false, false><<<dim3(HH / 128, NTOK / 128), 256, 0, stream>>>(
        Hb, wF2B, f2b, FF2, (bf16_t*)nullptr, NTOK, HH, FF);

    // 10) Residual + LN2 + 9-label projection -> out
    final_kernel<<<NTOK, 256, 0, stream>>>(FF2, LN1F, g2, bn2, ow, ob, out);
}
